// ExpertAttention_56590489092458
// MI455X (gfx1250) — matrix operand order repaired and checked by reading, not compiled, not run on hardware
//
#include <hip/hip_runtime.h>
#include <hip/hip_bf16.h>
#include <math.h>

// ---------------------------------------------------------------------------
// MI455X (gfx1250) routed MoE "ExpertAttention"
//   router GEMM  : v_wmma_f32_16x16x4_f32   (fp32 precision for top-k path)
//   expert FFN   : v_wmma_f32_16x16x32_bf16, double-buffered LDS pipeline,
//                  A tiles staged with global_load_async_to_lds_b128
//                  (ASYNCcnt) so activations bypass VGPRs entirely; B tiles
//                  fp32->bf16 converted in VGPRs, overlapped with WMMA.
// Sparse top-2 routing (compaction) cuts FFN FLOPs 4x vs dense reference.
// ---------------------------------------------------------------------------

typedef float   v2f   __attribute__((ext_vector_type(2)));
typedef float   v8f   __attribute__((ext_vector_type(8)));
typedef __bf16  v16bf __attribute__((ext_vector_type(16)));
typedef __bf16  v8bf  __attribute__((ext_vector_type(8)));
typedef unsigned short u16;
typedef unsigned int   u32;

#define N_TOK 4096
#define DIM   1024
#define HID   4096
#define NEXP  8
#define MAXD  16

// workspace byte offsets (needs ~121 MB of ws)
#define WS_Q    ((size_t)0)            // f32  N*DIM          16 MB
#define WS_XB   ((size_t)16u << 20)    // bf16 N*DIM           8 MB
#define WS_H2   ((size_t)24u << 20)    // bf16 (2N)*HID       64 MB
#define WS_OUTS ((size_t)88u << 20)    // f32  (2N)*DIM       32 MB
#define WS_META ((size_t)120u << 20)

struct Meta {
  int   cnt[NEXP];
  int   wp[NEXP];
  int   base[NEXP];
  int   tok_e[N_TOK * 2];
  float tok_g[N_TOK * 2];
  int   slotmap[N_TOK * 2];
  int   toklist[N_TOK * 2];
};

__device__ __forceinline__ u16 f2bf(float x) {           // RNE fp32 -> bf16
  u32 u = __float_as_uint(x);
  u += 0x7FFFu + ((u >> 16) & 1u);
  return (u16)(u >> 16);
}

// 32-byte async global->LDS copy (2 x B128, same operands, INST_OFFSET
// advances both LDS and global addresses per ISA 15.18 async pseudocode).
__device__ __forceinline__ void async_cp32(const u16* ldsDst, const u16* gSrc) {
  unsigned l = (unsigned)(unsigned long long)ldsDst;   // LDS aperture: low 32b
  unsigned long long g = (unsigned long long)gSrc;
  asm volatile("global_load_async_to_lds_b128 %0, %1, off"
               :: "v"(l), "v"(g) : "memory");
  asm volatile("global_load_async_to_lds_b128 %0, %1, off offset:16"
               :: "v"(l), "v"(g) : "memory");
}
__device__ __forceinline__ void wait_async0() {
  asm volatile("s_wait_asynccnt 0x0" ::: "memory");
}

// ------------------------------- zero meta ---------------------------------
__global__ void k_zero(Meta* meta) {
  int t = threadIdx.x;
  if (t < NEXP) { meta->cnt[t] = 0; meta->wp[t] = 0; }
}

// --------------------------- x -> bf16 convert -----------------------------
__global__ __launch_bounds__(256) void k_cvt_x(const float* __restrict__ x,
                                               u16* __restrict__ xb) {
  size_t i = ((size_t)blockIdx.x * blockDim.x + threadIdx.x) * 8;
  float4 f0 = *(const float4*)(x + i);
  float4 f1 = *(const float4*)(x + i + 4);
  uint4 o;
  o.x = f2bf(f0.x) | ((u32)f2bf(f0.y) << 16);
  o.y = f2bf(f0.z) | ((u32)f2bf(f0.w) << 16);
  o.z = f2bf(f1.x) | ((u32)f2bf(f1.y) << 16);
  o.w = f2bf(f1.z) | ((u32)f2bf(f1.w) << 16);
  *(uint4*)(xb + i) = o;
}

// --------------------------- router GEMM (f32 WMMA) ------------------------
__global__ __launch_bounds__(256) void k_router(const float* __restrict__ x,
                                                const float* __restrict__ wr,
                                                float* __restrict__ q) {
  const int tid = threadIdx.x, w = tid >> 5, lane = tid & 31;
  const int wm = w & 1, wn = w >> 1;
  const int m0 = blockIdx.x * 32 + wm * 16;
  const int n0 = blockIdx.y * 64 + wn * 16;
  const int r = lane & 15, hi = lane >> 4;
  const float* ap = x + (size_t)(m0 + r) * DIM + hi * 2;
  const float* bp = wr + (size_t)(n0 + r) * DIM + hi * 2;
  v8f acc = {};
#pragma unroll 8
  for (int k = 0; k < DIM; k += 4) {
    v2f a = *(const v2f*)(ap + k);
    v2f b = *(const v2f*)(bp + k);
    acc = __builtin_amdgcn_wmma_f32_16x16x4_f32(false, a, false, b,
                                                (short)0, acc, false, false);
  }
  float* qp = q + (size_t)(m0 + hi * 8) * DIM + n0 + r;
#pragma unroll
  for (int i = 0; i < 8; ++i) qp[(size_t)i * DIM] = acc[i];
}

// ----------------- gating: rmsnorm + rope + logits + top-2 -----------------
__global__ __launch_bounds__(256) void k_gate(const float* __restrict__ q,
                                              const float* __restrict__ keys,
                                              const float* __restrict__ bias,
                                              const int* __restrict__ depth_p,
                                              Meta* meta) {
  const int lane = threadIdx.x & 31;
  const int n = blockIdx.x * 8 + (threadIdx.x >> 5);
  const float* qr = q + (size_t)n * DIM;
  float qv[32];
  float ss = 0.f;
#pragma unroll
  for (int j = 0; j < 32; ++j) { float v = qr[j * 32 + lane]; qv[j] = v; ss += v * v; }
#pragma unroll
  for (int off = 16; off; off >>= 1) ss += __shfl_xor(ss, off);
  const float scale = rsqrtf(ss * (1.0f / DIM) + 1.1920929e-07f);
#pragma unroll
  for (int j = 0; j < 32; ++j) qv[j] *= scale;

  const float depth = (float)depth_p[0];
  const float rdep = (float)(MAXD - 1) - depth;
  const float lnb = 9.210340371976184f;          // ln(10000)
#pragma unroll
  for (int j = 0; j < 16; ++j) {
    int i = j * 32 + lane;                       // 0..511
    float invf = expf(-lnb * ((float)i * (1.0f / 512.0f)));
    float af = depth * invf, ar = rdep * invf;
    float q1 = qv[j], q2 = qv[j + 16];
    qv[j]      =  q1 * cosf(af) + q2 * sinf(af);
    qv[j + 16] = -q1 * sinf(ar) + q2 * cosf(ar);
  }
  float lg[NEXP];
#pragma unroll
  for (int e = 0; e < NEXP; ++e) {
    const float* kr = keys + e * DIM;
    float s = 0.f;
#pragma unroll
    for (int j = 0; j < 32; ++j) s += qv[j] * kr[j * 32 + lane];
#pragma unroll
    for (int off = 16; off; off >>= 1) s += __shfl_xor(s, off);
    lg[e] = s * (1.0f / 32.0f);                  // /sqrt(dim)
  }
  if (lane == 0) {
    int e0 = 0; float b0 = lg[0] + bias[0];
    for (int e = 1; e < NEXP; ++e) { float b = lg[e] + bias[e]; if (b > b0) { b0 = b; e0 = e; } }
    int e1 = -1; float b1 = -INFINITY;
    for (int e = 0; e < NEXP; ++e) {
      if (e == e0) continue;
      float b = lg[e] + bias[e]; if (b > b1) { b1 = b; e1 = e; }
    }
    float g0 = 1.f / (1.f + expf(-lg[e0]));
    float g1 = 1.f / (1.f + expf(-lg[e1]));
    float den = fmaxf(g0 + g1, 1e-9f);
    meta->tok_e[n * 2] = e0; meta->tok_e[n * 2 + 1] = e1;
    meta->tok_g[n * 2] = g0 / den; meta->tok_g[n * 2 + 1] = g1 / den;
    atomicAdd(&meta->cnt[e0], 1);
    atomicAdd(&meta->cnt[e1], 1);
  }
}

__global__ void k_prefix(Meta* meta) {
  if (threadIdx.x == 0) {
    int s = 0;
    for (int e = 0; e < NEXP; ++e) { meta->base[e] = s; s += meta->cnt[e]; meta->wp[e] = 0; }
  }
}

__global__ __launch_bounds__(256) void k_assign(Meta* meta) {
  int n = blockIdx.x * blockDim.x + threadIdx.x;
#pragma unroll
  for (int k2 = 0; k2 < 2; ++k2) {
    int e = meta->tok_e[n * 2 + k2];
    int pos = atomicAdd(&meta->wp[e], 1);
    int slot = meta->base[e] + pos;
    meta->toklist[slot] = n;
    meta->slotmap[n * 2 + k2] = slot;
  }
}

// ------------------- shared pieces of the two FFN GEMMs --------------------
#define LDA 40   // padded K stride (u16) for A tile
#define LDB 48   // padded K stride (u16) for B tile

__device__ __forceinline__ void load_b_regs(const float* bSrc, float4 fr[4]) {
  fr[0] = *(const float4*)(bSrc);
  fr[1] = *(const float4*)(bSrc + 4);
  fr[2] = *(const float4*)(bSrc + 8);
  fr[3] = *(const float4*)(bSrc + 12);
}
__device__ __forceinline__ void cvt_store_b(u16* dst, const float4 fr[4]) {
  uint4 o;
  o.x = f2bf(fr[0].x) | ((u32)f2bf(fr[0].y) << 16);
  o.y = f2bf(fr[0].z) | ((u32)f2bf(fr[0].w) << 16);
  o.z = f2bf(fr[1].x) | ((u32)f2bf(fr[1].y) << 16);
  o.w = f2bf(fr[1].z) | ((u32)f2bf(fr[1].w) << 16);
  *(uint4*)dst = o;
  o.x = f2bf(fr[2].x) | ((u32)f2bf(fr[2].y) << 16);
  o.y = f2bf(fr[2].z) | ((u32)f2bf(fr[2].w) << 16);
  o.z = f2bf(fr[3].x) | ((u32)f2bf(fr[3].y) << 16);
  o.w = f2bf(fr[3].z) | ((u32)f2bf(fr[3].w) << 16);
  *(uint4*)(dst + 8) = o;
}
__device__ __forceinline__ void wmma_tile(const u16* sA, const u16* sB,
                                          int wm, int wn, int r, int akb, int bkb,
                                          v8f acc[4][2]) {
  v16bf bfr[2];
#pragma unroll
  for (int ni = 0; ni < 2; ++ni)
    bfr[ni] = *(const v16bf*)&sB[(size_t)(wn * 32 + ni * 16 + r) * LDB + bkb];
#pragma unroll
  for (int mi = 0; mi < 4; ++mi) {
    const u16* ab = &sA[(size_t)(wm * 64 + mi * 16 + r) * LDA + akb];
    v16bf a;
    *(v8bf*)&a       = *(const v8bf*)(ab);
    *((v8bf*)&a + 1) = *(const v8bf*)(ab + 8);
#pragma unroll
    for (int ni = 0; ni < 2; ++ni)
      acc[mi][ni] = __builtin_amdgcn_wmma_f32_16x16x32_bf16(
          false, a, false, bfr[ni], (short)0, acc[mi][ni], false, false);
  }
}

// ---------------------- FFN pass 1: h2 = relu(X fc^T)^2 --------------------
// 128x128 tile per block; double-buffered LDS; A tile via async-to-LDS.
__global__ __launch_bounds__(256) void k_ffn1(const u16* __restrict__ xb,
                                              const float* __restrict__ fcw,
                                              const Meta* __restrict__ meta,
                                              u16* __restrict__ h2) {
  __shared__ __align__(64) u16 sA[2][128 * LDA];
  __shared__ __align__(64) u16 sB[2][128 * LDB];
  const int e = blockIdx.x >> 5;
  const int t = blockIdx.x & 31;
  const int cnt = meta->cnt[e];
  if (t * 128 >= cnt) return;                       // uniform early-exit
  const int baseSlot = meta->base[e] + t * 128;
  const int h0 = blockIdx.y * 128;

  const int tid = threadIdx.x;
  const int lrow = tid >> 1, lhalf = tid & 1;
  const int tok = (t * 128 + lrow < cnt) ? meta->toklist[baseSlot + lrow] : 0;
  const u16* aSrc = xb + (size_t)tok * DIM + lhalf * 16;
  const float* bSrc = fcw + ((size_t)e * HID + h0 + lrow) * DIM + lhalf * 16;
  u16* const aDst0 = &sA[0][lrow * LDA + lhalf * 16];
  u16* const aDst1 = &sA[1][lrow * LDA + lhalf * 16];
  u16* const bDst0 = &sB[0][lrow * LDB + lhalf * 16];
  u16* const bDst1 = &sB[1][lrow * LDB + lhalf * 16];

  const int w = tid >> 5, lane = tid & 31;
  const int wm = w & 1, wn = w >> 1;
  const int r = lane & 15, hih = lane >> 4;
  const int akb = hih * 16, bkb = hih * 16;
  v8f acc[4][2] = {};

  // prologue: stage k0 = 0 into buffer 0
  async_cp32(aDst0, aSrc);
  float4 fr[4];
  load_b_regs(bSrc, fr);
  cvt_store_b(bDst0, fr);
  wait_async0();
  __syncthreads();

  const int KITER = DIM / 32;
  int buf = 0;
  for (int i = 0; i < KITER; ++i) {
    const int kn = (i + 1) * 32;
    if (i + 1 < KITER) {                            // stage next tile
      async_cp32(buf ? aDst0 : aDst1, aSrc + kn);
      load_b_regs(bSrc + kn, fr);                   // latency hidden by WMMA
    }
    wmma_tile(sA[buf], sB[buf], wm, wn, r, akb, bkb, acc);
    if (i + 1 < KITER) cvt_store_b(buf ? bDst0 : bDst1, fr);
    wait_async0();
    __syncthreads();
    buf ^= 1;
  }
  // epilogue: relu^2 -> bf16
#pragma unroll
  for (int mi = 0; mi < 4; ++mi) {
    const int mrow = wm * 64 + mi * 16 + hih * 8;
#pragma unroll
    for (int ni = 0; ni < 2; ++ni) {
      const int col = h0 + wn * 32 + ni * 16 + r;
#pragma unroll
      for (int i = 0; i < 8; ++i) {
        int m = mrow + i;
        if (t * 128 + m < cnt) {
          float v = fmaxf(acc[mi][ni][i], 0.f);
          h2[(size_t)(baseSlot + m) * HID + col] = f2bf(v * v);
        }
      }
    }
  }
}

// ---------------------- FFN pass 2: outs = h2 proj^T -----------------------
__global__ __launch_bounds__(256) void k_ffn2(const u16* __restrict__ h2,
                                              const float* __restrict__ pw,
                                              const Meta* __restrict__ meta,
                                              float* __restrict__ outs) {
  __shared__ __align__(64) u16 sA[2][128 * LDA];
  __shared__ __align__(64) u16 sB[2][128 * LDB];
  const int e = blockIdx.x >> 5;
  const int t = blockIdx.x & 31;
  const int cnt = meta->cnt[e];
  if (t * 128 >= cnt) return;
  const int baseSlot = meta->base[e] + t * 128;
  const int d0 = blockIdx.y * 128;

  const int tid = threadIdx.x;
  const int lrow = tid >> 1, lhalf = tid & 1;
  const u16* aSrc = h2 + (size_t)(baseSlot + lrow) * HID + lhalf * 16;
  const float* bSrc = pw + ((size_t)e * DIM + d0 + lrow) * HID + lhalf * 16;
  u16* const aDst0 = &sA[0][lrow * LDA + lhalf * 16];
  u16* const aDst1 = &sA[1][lrow * LDA + lhalf * 16];
  u16* const bDst0 = &sB[0][lrow * LDB + lhalf * 16];
  u16* const bDst1 = &sB[1][lrow * LDB + lhalf * 16];

  const int w = tid >> 5, lane = tid & 31;
  const int wm = w & 1, wn = w >> 1;
  const int r = lane & 15, hih = lane >> 4;
  const int akb = hih * 16, bkb = hih * 16;
  v8f acc[4][2] = {};

  async_cp32(aDst0, aSrc);
  float4 fr[4];
  load_b_regs(bSrc, fr);
  cvt_store_b(bDst0, fr);
  wait_async0();
  __syncthreads();

  const int KITER = HID / 32;
  int buf = 0;
  for (int i = 0; i < KITER; ++i) {
    const int kn = (i + 1) * 32;
    if (i + 1 < KITER) {
      async_cp32(buf ? aDst0 : aDst1, aSrc + kn);
      load_b_regs(bSrc + kn, fr);
    }
    wmma_tile(sA[buf], sB[buf], wm, wn, r, akb, bkb, acc);
    if (i + 1 < KITER) cvt_store_b(buf ? bDst0 : bDst1, fr);
    wait_async0();
    __syncthreads();
    buf ^= 1;
  }
#pragma unroll
  for (int mi = 0; mi < 4; ++mi) {
    const int mrow = wm * 64 + mi * 16 + hih * 8;
#pragma unroll
    for (int ni = 0; ni < 2; ++ni) {
      const int col = d0 + wn * 32 + ni * 16 + r;
#pragma unroll
      for (int i = 0; i < 8; ++i) {
        int m = mrow + i;
        if (t * 128 + m < cnt)
          outs[(size_t)(baseSlot + m) * DIM + col] = acc[mi][ni][i];
      }
    }
  }
}

// ----------------------- combine: y = g0*out0 + g1*out1 --------------------
__global__ __launch_bounds__(256) void k_combine(const float* __restrict__ outs,
                                                 const Meta* __restrict__ meta,
                                                 float* __restrict__ y) {
  int idx = blockIdx.x * blockDim.x + threadIdx.x;   // over N*DIM/4
  int n = idx >> 8;                                   // DIM/4 == 256
  int d4 = (idx & 255) * 4;
  int s0 = meta->slotmap[n * 2], s1 = meta->slotmap[n * 2 + 1];
  float g0 = meta->tok_g[n * 2], g1 = meta->tok_g[n * 2 + 1];
  float4 a = *(const float4*)(outs + (size_t)s0 * DIM + d4);
  float4 b = *(const float4*)(outs + (size_t)s1 * DIM + d4);
  float4 o;
  o.x = g0 * a.x + g1 * b.x;
  o.y = g0 * a.y + g1 * b.y;
  o.z = g0 * a.z + g1 * b.z;
  o.w = g0 * a.w + g1 * b.w;
  *(float4*)(y + (size_t)n * DIM + d4) = o;
}

// ---------------------------------------------------------------------------
extern "C" void kernel_launch(void* const* d_in, const int* in_sizes, int n_in,
                              void* d_out, int out_size, void* d_ws, size_t ws_size,
                              hipStream_t stream) {
  const float* x    = (const float*)d_in[0];
  const float* wr   = (const float*)d_in[1];
  const float* keys = (const float*)d_in[2];
  const float* bias = (const float*)d_in[3];
  const float* fcw  = (const float*)d_in[4];
  const float* pw   = (const float*)d_in[5];
  const int*   dpt  = (const int*)d_in[6];
  float* y = (float*)d_out;

  char* ws = (char*)d_ws;
  float* q    = (float*)(ws + WS_Q);
  u16*   xb   = (u16*)(ws + WS_XB);
  u16*   h2   = (u16*)(ws + WS_H2);
  float* outs = (float*)(ws + WS_OUTS);
  Meta*  meta = (Meta*)(ws + WS_META);

  k_zero<<<1, 32, 0, stream>>>(meta);
  k_cvt_x<<<(N_TOK * DIM) / (256 * 8), 256, 0, stream>>>(x, xb);
  k_router<<<dim3(N_TOK / 32, DIM / 64), 256, 0, stream>>>(x, wr, q);
  k_gate<<<N_TOK / 8, 256, 0, stream>>>(q, keys, bias, dpt, meta);
  k_prefix<<<1, 32, 0, stream>>>(meta);
  k_assign<<<N_TOK / 256, 256, 0, stream>>>(meta);
  k_ffn1<<<dim3(NEXP * 32, HID / 128), 256, 0, stream>>>(xb, fcw, meta, h2);
  k_ffn2<<<dim3(NEXP * 32, DIM / 128), 256, 0, stream>>>(h2, pw, meta, outs);
  k_combine<<<(N_TOK * DIM / 4) / 256, 256, 0, stream>>>(outs, meta, y);
}
